// Attention_86698209837155
// MI455X (gfx1250) — matrix operand order repaired and checked by reading, not compiled, not run on hardware
//
#include <hip/hip_runtime.h>
#include <hip/hip_bf16.h>
#include <float.h>

// ---------------------------------------------------------------------------
// MI455X (gfx1250) implementation — round 3.
//   K0  prep        : x -> nan_to_num -> feature(fp32 out) + bf16 shadow (ws, L2-resident)
//   K1  wconv       : Wv/Wu/seg_W1 -> bf16
//   K2  attn_seg    : 64-rows-per-block fused 3-GEMM, 12 v_wmma per k-step per wave,
//                     A-tile staged by the TDM (tensor_load_to_lds + s_wait_tensorcnt),
//                     deterministic epilogue (no float atomics)
//   K3  softmax/max : per-batch softmax over A, max over output_seg
//   K4  bag         : softmax-weighted feature reduction (bf16, L2 hits)
//   K5a/b bag MLP   : L1 spread over 32 blocks, tail + output assembly in one block
//   K6a-d kmeans2   : whole-GPU iterative Lloyd (78 launches in the graph):
//                     init -> 26x[prep, assign] + 25x[update]
// ---------------------------------------------------------------------------

typedef __attribute__((ext_vector_type(16))) __bf16 v16bf;
typedef __attribute__((ext_vector_type(8)))  float  v8f;
typedef __attribute__((ext_vector_type(4)))  unsigned int u32x4;
typedef __attribute__((ext_vector_type(8)))  int          i32x8;
typedef __attribute__((ext_vector_type(4)))  int          i32x4;

#define Bb   16
#define Tt   1024
#define Dd   4096
#define N1   256
#define N2   128
#define N3   32
#define ROWS (Bb * Tt)          // 16384
#define MROW 64                 // rows per K2 block

// workspace byte offsets
#define OFF_FEATBF  ((size_t)0)                         // 134217728
#define OFF_WV      ((size_t)134217728)
#define OFF_WU      (OFF_WV + (size_t)2097152)
#define OFF_W1      (OFF_WU + (size_t)2097152)
#define OFF_A       (OFF_W1 + (size_t)2097152)          // 65536
#define OFF_WSM     (OFF_A  + (size_t)65536)            // 65536
#define OFF_BAG     (OFF_WSM + (size_t)65536)           // 262144
#define OFF_O2      (OFF_BAG + (size_t)262144)          // 256 (padded)
#define OFF_CENT    (OFF_O2  + (size_t)256)             // 16*2*4096*4 = 524288
#define OFF_LBL     (OFF_CENT + (size_t)524288)         // 16*1024*4   = 65536
#define OFF_CNT     (OFF_LBL + (size_t)65536)           // 256 (padded)
#define OFF_C2      (OFF_CNT + (size_t)256)             // 256 (padded)
#define OFF_H1B     (OFF_C2  + (size_t)256)             // 16*256*4 = 16384

// d_out float-element offsets (tuple flattened in return order)
#define O_FEAT ((size_t)0)
#define O_LBL  ((size_t)67108864)
#define O_SEG  (O_LBL + (size_t)16384)
#define O_OUT  (O_SEG + (size_t)16384)
#define O_AT   (O_OUT + (size_t)32)

__device__ __forceinline__ float sigmoidf(float x) { return 1.0f / (1.0f + __expf(-x)); }

__device__ __forceinline__ unsigned pack_bf2(float a, float b) {
    union { __bf16 h; unsigned short u; } ca, cb;
    ca.h = (__bf16)a; cb.h = (__bf16)b;
    return (unsigned)ca.u | ((unsigned)cb.u << 16);
}

// ---------------------------------------------------------------------------
// K0: nan_to_num, fp32 feature out, bf16 shadow copy to ws.
// ---------------------------------------------------------------------------
__global__ void k0_prep(const float* __restrict__ x, float* __restrict__ featF,
                        __bf16* __restrict__ featH) {
    const size_t row = blockIdx.x;
    const size_t base = row * (size_t)Dd + (size_t)threadIdx.x * 16;
    #pragma unroll
    for (int c = 0; c < 4; ++c) {
        float4 v = *reinterpret_cast<const float4*>(x + base + c * 4);
        float f[4] = {v.x, v.y, v.z, v.w};
        #pragma unroll
        for (int e = 0; e < 4; ++e) {
            float t = f[e];
            if (t != t) t = 0.0f;                            // NaN -> 0
            t = fminf(fmaxf(t, -FLT_MAX), FLT_MAX);          // +/-inf -> +/-max
            f[e] = t;
        }
        *reinterpret_cast<float4*>(featF + base + c * 4) = make_float4(f[0], f[1], f[2], f[3]);
        uint2 p = make_uint2(pack_bf2(f[0], f[1]), pack_bf2(f[2], f[3]));
        *reinterpret_cast<uint2*>(featH + base + c * 4) = p;
    }
}

// ---------------------------------------------------------------------------
// K1: convert three 4096x256 weight matrices to bf16.
// ---------------------------------------------------------------------------
__global__ void k1_wconv(const float* __restrict__ wv, const float* __restrict__ wu,
                         const float* __restrict__ w1, __bf16* __restrict__ hv,
                         __bf16* __restrict__ hu, __bf16* __restrict__ h1) {
    size_t i = ((size_t)blockIdx.x * blockDim.x + threadIdx.x) * 4;
    #pragma unroll
    for (int e = 0; e < 4; ++e) {
        hv[i + e] = (__bf16)wv[i + e];
        hu[i + e] = (__bf16)wu[i + e];
        h1[i + e] = (__bf16)w1[i + e];
    }
}

// ---------------------------------------------------------------------------
// K2: fused attention + seg-classifier kernel, 64 rows / block.
// ---------------------------------------------------------------------------
__global__ void __launch_bounds__(512)
k2_attn_seg(const __bf16* __restrict__ featH,
            const __bf16* __restrict__ wvH, const __bf16* __restrict__ wuH,
            const __bf16* __restrict__ w1H,
            const float* __restrict__ bv, const float* __restrict__ bu,
            const float* __restrict__ Wg, const float* __restrict__ bg,
            const float* __restrict__ sb1,
            const float* __restrict__ sW2, const float* __restrict__ sb2,
            const float* __restrict__ sW3, const float* __restrict__ sb3,
            const float* __restrict__ sW4, const float* __restrict__ sb4,
            float* __restrict__ segOut, float* __restrict__ AtOut,
            float* __restrict__ wsA) {
    __shared__ __align__(16) unsigned char smem[61952];
    // phase 1 carve
    __bf16* Asm = reinterpret_cast<__bf16*>(smem);                 // [64][32]   4KB
    __bf16* Bv  = reinterpret_cast<__bf16*>(smem + 4096);          // [256][32] 16KB
    __bf16* Bu  = reinterpret_cast<__bf16*>(smem + 20480);
    __bf16* Bw  = reinterpret_cast<__bf16*>(smem + 36864);         // ends 53248
    // phase 2 carve (aliases phase 1 after the K loop)
    __bf16* H1h = reinterpret_cast<__bf16*>(smem);                 // [64][256] 32KB
    __bf16* H2h = reinterpret_cast<__bf16*>(smem + 32768);         // [64][128] 16KB
    float*  H3  = reinterpret_cast<float*>(smem + 49152);          // [64][32]   8KB
    float*  Apart = reinterpret_cast<float*>(smem + 57344);        // [16][64]   4KB
    float*  Aval  = reinterpret_cast<float*>(smem + 61440);        // [64]

    const int tid  = threadIdx.x;
    const int wave = tid >> 5;
    const int lane = tid & 31;
    const int row0 = blockIdx.x * MROW;

    v8f accV[4] = {}, accU[4] = {}, accW[4] = {};

    // fragment coordinates (contiguous k: element s of lane half h holds k = 16h + s)
    const int mA  = lane & 15;                 // A frag row within 16-row tile
    const int kbA = (lane >> 4) * 16;          // A frag K sub-base
    const int nB  = wave * 16 + (lane & 15);   // B frag column
    const int kbB = (lane >> 4) * 16;          // B frag K sub-base

    // B staging coordinates: thread covers 8 consecutive n at fixed k (x2 k's)
    const int kS  = tid >> 5;                  // 0..15  (+16 on second chunk)
    const int n0S = (tid & 31) * 8;            // 0..248
    const size_t gOffBase = (size_t)kS * N1 + n0S;

    // TDM descriptor for the A tile: 2D, bf16, tile 32(K) x 64(rows)
    const unsigned ldsA = (unsigned)(size_t)(void*)Asm;
    const size_t gaBase = (size_t)featH + (size_t)row0 * Dd * 2;
    const i32x8 g1 = { 0x00010000,   // data_size=2B
                       0x10000000,   // tensor_dim0 = 4096 (lo16 << 16)
                       0x00400000,   // tensor_dim1 = 64   (lo16 << 16)
                       0x00200000,   // tile_dim0   = 32   (<<16)
                       64,           // tile_dim1   = 64
                       4096,         // tensor_dim0_stride lo
                       0, 0 };
    const i32x4 z4 = {0, 0, 0, 0};

    for (int k0 = 0; k0 < Dd; k0 += 32) {
        // --- issue TDM for the A tile (one wave; EXEC is ignored by TDM) ---
        if (wave == 0) {
            size_t ga = gaBase + (size_t)k0 * 2;
            u32x4 g0;
            g0.x = 1u;                                   // count=1, user mode
            g0.y = ldsA;                                 // lds_addr
            g0.z = (unsigned)(ga & 0xFFFFFFFFu);
            g0.w = (unsigned)((ga >> 32) & 0x01FFFFFFu) | (2u << 30);  // type=2
#if defined(__clang_major__) && (__clang_major__ >= 23)
            i32x8 z8 = {0, 0, 0, 0, 0, 0, 0, 0};
            __builtin_amdgcn_tensor_load_to_lds(g0, g1, z4, z4, z8, 0);
#else
            __builtin_amdgcn_tensor_load_to_lds(g0, g1, z4, z4, 0);
#endif
        }
        // --- stage B tiles (transposed to N-major), vectorized loads ---
        #pragma unroll
        for (int c = 0; c < 2; ++c) {
            size_t go = gOffBase + (size_t)(k0 + 16 * c) * N1;
            uint4 pv = *reinterpret_cast<const uint4*>(wvH + go);
            uint4 pu = *reinterpret_cast<const uint4*>(wuH + go);
            uint4 pw = *reinterpret_cast<const uint4*>(w1H + go);
            const __bf16* ev = reinterpret_cast<const __bf16*>(&pv);
            const __bf16* eu = reinterpret_cast<const __bf16*>(&pu);
            const __bf16* ew = reinterpret_cast<const __bf16*>(&pw);
            const int l0 = n0S * 32 + kS + 16 * c;
            #pragma unroll
            for (int e = 0; e < 8; ++e) {
                Bv[l0 + e * 32] = ev[e];
                Bu[l0 + e * 32] = eu[e];
                Bw[l0 + e * 32] = ew[e];
            }
        }
        if (k0 + 32 < Dd) {
            __builtin_prefetch(&wvH[(size_t)(k0 + 32) * N1], 0, 1);
            __builtin_prefetch(&wuH[(size_t)(k0 + 32) * N1], 0, 1);
            __builtin_prefetch(&w1H[(size_t)(k0 + 32) * N1], 0, 1);
        }
        if (wave == 0) __builtin_amdgcn_s_wait_tensorcnt(0);
        __syncthreads();

        // --- fragments + 12 WMMAs ---
        v16bf b0, b1, b2;
        {
            uint4* p = reinterpret_cast<uint4*>(&b0);
            p[0] = *reinterpret_cast<const uint4*>(&Bv[nB * 32 + kbB]);
            p[1] = *reinterpret_cast<const uint4*>(&Bv[nB * 32 + kbB + 8]);
        }
        {
            uint4* p = reinterpret_cast<uint4*>(&b1);
            p[0] = *reinterpret_cast<const uint4*>(&Bu[nB * 32 + kbB]);
            p[1] = *reinterpret_cast<const uint4*>(&Bu[nB * 32 + kbB + 8]);
        }
        {
            uint4* p = reinterpret_cast<uint4*>(&b2);
            p[0] = *reinterpret_cast<const uint4*>(&Bw[nB * 32 + kbB]);
            p[1] = *reinterpret_cast<const uint4*>(&Bw[nB * 32 + kbB + 8]);
        }
        #pragma unroll
        for (int rt = 0; rt < 4; ++rt) {
            v16bf a;
            uint4* p = reinterpret_cast<uint4*>(&a);
            const int ar = (rt * 16 + mA) * 32 + kbA;
            p[0] = *reinterpret_cast<const uint4*>(&Asm[ar]);
            p[1] = *reinterpret_cast<const uint4*>(&Asm[ar + 8]);
            accV[rt] = __builtin_amdgcn_wmma_f32_16x16x32_bf16(false, a, false, b0,
                                                               (short)0, accV[rt], false, false);
            accU[rt] = __builtin_amdgcn_wmma_f32_16x16x32_bf16(false, a, false, b1,
                                                               (short)0, accU[rt], false, false);
            accW[rt] = __builtin_amdgcn_wmma_f32_16x16x32_bf16(false, a, false, b2,
                                                               (short)0, accW[rt], false, false);
        }
        __syncthreads();
    }

    // ---- epilogue: gated attention score + seg L1 (deterministic reduce) ----
    const float bvn = bv[nB], bun = bu[nB], wgn = Wg[nB], sb1n = sb1[nB];
    const int mhalf = 8 * (lane >> 4);
    #pragma unroll
    for (int rt = 0; rt < 4; ++rt) {
        #pragma unroll
        for (int r = 0; r < 8; ++r) {
            int m = rt * 16 + r + mhalf;
            float av = tanhf(accV[rt][r] + bvn);
            float au = sigmoidf(accU[rt][r] + bun);
            float p  = av * au * wgn;                       // partial of A[m]
            for (int o = 8; o > 0; o >>= 1) p += __shfl_xor(p, o, 16);
            if ((lane & 15) == 0) Apart[wave * MROW + m] = p;
            H1h[m * N1 + nB] = (__bf16)fmaxf(accW[rt][r] + sb1n, 0.0f);
        }
    }
    __syncthreads();
    if (tid < MROW) {
        float A = bg[0];
        for (int w = 0; w < 16; ++w) A += Apart[w * MROW + tid];   // fixed order
        Aval[tid] = A;
        int rg = row0 + tid;
        AtOut[rg] = A;
        wsA[rg]   = A;
    }
    __syncthreads();

    // seg L2: [64,256] @ [256,128]
    for (int e = tid; e < MROW * N2; e += 512) {
        int m = e >> 7, j = e & (N2 - 1);
        float s = sb2[j];
        for (int i = 0; i < N1; ++i) s += (float)H1h[m * N1 + i] * sW2[i * N2 + j];
        H2h[m * N2 + j] = (__bf16)fmaxf(s, 0.0f);
    }
    __syncthreads();
    // seg L3: [64,128] @ [128,32]
    for (int e = tid; e < MROW * N3; e += 512) {
        int m = e >> 5, j = e & 31;
        float s = sb3[j];
        for (int i = 0; i < N2; ++i) s += (float)H2h[m * N2 + i] * sW3[i * N3 + j];
        H3[m * N3 + j] = fmaxf(s, 0.0f);
    }
    __syncthreads();
    // seg L4 + boost + sigmoid
    if (tid < MROW) {
        int m = tid;
        float s = sb4[0];
        for (int i = 0; i < N3; ++i) s += H3[m * N3 + i] * sW4[i];
        float boost = 2.0f * sigmoidf(Aval[m]);
        segOut[row0 + m] = sigmoidf(s * boost);
    }
}

// ---------------------------------------------------------------------------
// reduction helpers (1024-thread blocks)
// ---------------------------------------------------------------------------
__device__ __forceinline__ float block_max(float v) {
    __shared__ float red[32];
    __shared__ float bc;
    int lane = threadIdx.x & 31, w = threadIdx.x >> 5;
    for (int o = 16; o > 0; o >>= 1) v = fmaxf(v, __shfl_xor(v, o, 32));
    __syncthreads();
    if (lane == 0) red[w] = v;
    __syncthreads();
    if (w == 0) {
        float r = red[lane];
        for (int o = 16; o > 0; o >>= 1) r = fmaxf(r, __shfl_xor(r, o, 32));
        if (lane == 0) bc = r;
    }
    __syncthreads();
    return bc;
}

__device__ __forceinline__ float block_sum(float v) {
    __shared__ float red[32];
    __shared__ float bc;
    int lane = threadIdx.x & 31, w = threadIdx.x >> 5;
    for (int o = 16; o > 0; o >>= 1) v += __shfl_xor(v, o, 32);
    __syncthreads();
    if (lane == 0) red[w] = v;
    __syncthreads();
    if (w == 0) {
        float r = red[lane];
        for (int o = 16; o > 0; o >>= 1) r += __shfl_xor(r, o, 32);
        if (lane == 0) bc = r;
    }
    __syncthreads();
    return bc;
}

// ---------------------------------------------------------------------------
// K3: per-batch softmax over A and max over output_seg
// ---------------------------------------------------------------------------
__global__ void __launch_bounds__(1024)
k3_softmax_max(const float* __restrict__ wsA, const float* __restrict__ segOut,
               float* __restrict__ wsW, float* __restrict__ wsO2) {
    int b = blockIdx.x, t = threadIdx.x;
    float a = wsA[b * Tt + t];
    float mx = block_max(a);
    float e = __expf(a - mx);
    float s = block_sum(e);
    wsW[b * Tt + t] = e / s;
    float sm = block_max(segOut[b * Tt + t]);
    if (t == 0) wsO2[b] = sm;
}

// ---------------------------------------------------------------------------
// K4: bag[b][d] = sum_t w[b][t] * feature[b][t][d]
// ---------------------------------------------------------------------------
__global__ void __launch_bounds__(512)
k4_bag(const __bf16* __restrict__ featH, const float* __restrict__ wsW,
       float* __restrict__ wsBag) {
    __shared__ float wsh[Tt];
    int b = blockIdx.y;
    int d = blockIdx.x * 512 + threadIdx.x;
    for (int i = threadIdx.x; i < Tt; i += 512) wsh[i] = wsW[b * Tt + i];
    __syncthreads();
    float acc = 0.0f;
    const __bf16* fb = featH + (size_t)b * Tt * Dd + d;
    for (int t = 0; t < Tt; ++t)
        acc += wsh[t] * (float)fb[(size_t)t * Dd];
    wsBag[b * Dd + d] = acc;
}

// ---------------------------------------------------------------------------
// K5a: bag classifier layer 1, spread over (2 col-halves x 16 rows) blocks.
// ---------------------------------------------------------------------------
__global__ void __launch_bounds__(128)
k5a_bagL1(const float* __restrict__ wsBag, const float* __restrict__ W1,
          const float* __restrict__ b1, float* __restrict__ wsH1) {
    int m = blockIdx.y;
    int j = blockIdx.x * 128 + threadIdx.x;
    float s = b1[j];
    const float* br = wsBag + (size_t)m * Dd;
    for (int i = 0; i < Dd; ++i) s += br[i] * W1[(size_t)i * N1 + j];
    wsH1[m * N1 + j] = fmaxf(s, 0.0f);
}

// ---------------------------------------------------------------------------
// K5b: bag classifier tail (256->128->32->1) + final output assembly.
// ---------------------------------------------------------------------------
__global__ void __launch_bounds__(512)
k5b_bagtail(const float* __restrict__ wsH1,
            const float* __restrict__ W2, const float* __restrict__ b2,
            const float* __restrict__ W3, const float* __restrict__ b3,
            const float* __restrict__ W4, const float* __restrict__ b4,
            const float* __restrict__ wsO2, float* __restrict__ out) {
    __shared__ float H1[16 * N1];
    __shared__ float H2[16 * N2];
    __shared__ float H3[16 * N3];
    int tid = threadIdx.x;
    for (int e = tid; e < 16 * N1; e += 512) H1[e] = wsH1[e];
    __syncthreads();
    for (int e = tid; e < 16 * N2; e += 512) {
        int m = e >> 7, j = e & (N2 - 1);
        float s = b2[j];
        for (int i = 0; i < N1; ++i) s += H1[m * N1 + i] * W2[i * N2 + j];
        H2[m * N2 + j] = fmaxf(s, 0.0f);
    }
    __syncthreads();
    if (tid < 16 * N3) {
        int m = tid >> 5, j = tid & 31;
        float s = b3[j];
        for (int i = 0; i < N2; ++i) s += H2[m * N2 + i] * W3[i * N3 + j];
        H3[m * N3 + j] = fmaxf(s, 0.0f);
    }
    __syncthreads();
    if (tid < 16) {
        int m = tid;
        float s = b4[0];
        for (int i = 0; i < N3; ++i) s += H3[m * N3 + i] * W4[i];
        out[m * 2 + 0] = 0.5f * sigmoidf(s);
        out[m * 2 + 1] = 0.5f * wsO2[m];
    }
}

// ---------------------------------------------------------------------------
// K6a: kmeans init — centroids = f[0], f[T/2] per batch.
// ---------------------------------------------------------------------------
__global__ void __launch_bounds__(512)
k6a_init(const __bf16* __restrict__ featH, float* __restrict__ cent) {
    int b = blockIdx.x;
    const __bf16* fb = featH + (size_t)b * Tt * Dd;
    float* cb = cent + (size_t)b * 2 * Dd;
    for (int d = threadIdx.x; d < Dd; d += 512) {
        cb[d]      = (float)fb[d];
        cb[Dd + d] = (float)fb[(size_t)(Tt / 2) * Dd + d];
    }
}

// ---------------------------------------------------------------------------
// K6b: per-iteration prep — centroid norms + zero counts.
// ---------------------------------------------------------------------------
__global__ void __launch_bounds__(256)
k6b_prep(const float* __restrict__ cent, float* __restrict__ c2, int* __restrict__ cnt) {
    __shared__ float r0[8], r1[8];
    int b = blockIdx.x, tid = threadIdx.x;
    int lane = tid & 31, w = tid >> 5;
    const float* cb = cent + (size_t)b * 2 * Dd;
    float p0 = 0.0f, p1 = 0.0f;
    for (int d = tid; d < Dd; d += 256) {
        p0 += cb[d] * cb[d];
        p1 += cb[Dd + d] * cb[Dd + d];
    }
    for (int o = 16; o > 0; o >>= 1) {
        p0 += __shfl_xor(p0, o, 32);
        p1 += __shfl_xor(p1, o, 32);
    }
    if (lane == 0) { r0[w] = p0; r1[w] = p1; }
    __syncthreads();
    if (tid == 0) {
        float s0 = 0.0f, s1 = 0.0f;
        for (int i = 0; i < 8; ++i) { s0 += r0[i]; s1 += r1[i]; }   // fixed order
        c2[b * 2 + 0] = s0;
        c2[b * 2 + 1] = s1;
        cnt[b * 2 + 0] = 0;
        cnt[b * 2 + 1] = 0;
    }
}

// ---------------------------------------------------------------------------
// K6c: assignment — grid (32 point-chunks, 16 batches) x 256 threads.
// One wave per point, lane-strided dims, shuffle reduce. Also accumulates
// cluster counts (integer atomics = deterministic) and writes float labels.
// ---------------------------------------------------------------------------
__global__ void __launch_bounds__(256)
k6c_assign(const __bf16* __restrict__ featH, const float* __restrict__ cent,
           const float* __restrict__ c2, int* __restrict__ lbl,
           int* __restrict__ cnt, float* __restrict__ lblOutF) {
    int b = blockIdx.y;
    int wave = threadIdx.x >> 5, lane = threadIdx.x & 31;
    const float* c0p = cent + (size_t)b * 2 * Dd;
    const float* c1p = c0p + Dd;
    float e0b = c2[b * 2 + 0], e1b = c2[b * 2 + 1];
    #pragma unroll
    for (int i = 0; i < 4; ++i) {
        int t = blockIdx.x * 32 + i * 8 + wave;
        const __bf16* fr = featH + ((size_t)b * Tt + t) * Dd;
        float d0 = 0.0f, d1 = 0.0f;
        for (int d = lane; d < Dd; d += 32) {
            float f = (float)fr[d];
            d0 += f * c0p[d];
            d1 += f * c1p[d];
        }
        for (int o = 16; o > 0; o >>= 1) {
            d0 += __shfl_xor(d0, o, 32);
            d1 += __shfl_xor(d1, o, 32);
        }
        if (lane == 0) {
            int l = ((e1b - 2.0f * d1) < (e0b - 2.0f * d0)) ? 1 : 0;  // argmin, ties->0
            lbl[b * Tt + t] = l;
            lblOutF[b * Tt + t] = (float)l;
            atomicAdd(&cnt[b * 2 + l], 1);
        }
    }
}

// ---------------------------------------------------------------------------
// K6d: centroid update — grid (8 dim-chunks, 16 batches) x 512 threads.
// Thread-per-dim column scan; labels staged in LDS; deterministic sums.
// ---------------------------------------------------------------------------
__global__ void __launch_bounds__(512)
k6d_update(const __bf16* __restrict__ featH, const int* __restrict__ lbl,
           const int* __restrict__ cnt, float* __restrict__ cent) {
    __shared__ int ll[Tt];
    int b = blockIdx.y;
    int d = blockIdx.x * 512 + threadIdx.x;
    for (int i = threadIdx.x; i < Tt; i += 512) ll[i] = lbl[b * Tt + i];
    __syncthreads();
    const __bf16* fb = featH + (size_t)b * Tt * Dd;
    float s0 = 0.0f, s1 = 0.0f;
    for (int t = 0; t < Tt; ++t) {
        float f = (float)fb[(size_t)t * Dd + d];
        if (ll[t]) s1 += f; else s0 += f;
    }
    int n0 = cnt[b * 2 + 0], n1 = cnt[b * 2 + 1];
    float* cb = cent + (size_t)b * 2 * Dd;
    cb[d]      = (n0 > 0) ? s0 / (float)max(n0, 1) : cb[d];
    cb[Dd + d] = (n1 > 0) ? s1 / (float)max(n1, 1) : cb[Dd + d];
}

// ---------------------------------------------------------------------------
extern "C" void kernel_launch(void* const* d_in, const int* in_sizes, int n_in,
                              void* d_out, int out_size, void* d_ws, size_t ws_size,
                              hipStream_t stream) {
    const float* x   = (const float*)d_in[0];
    const float* Wv  = (const float*)d_in[1];
    const float* bv  = (const float*)d_in[2];
    const float* Wu  = (const float*)d_in[3];
    const float* bu  = (const float*)d_in[4];
    const float* Wg  = (const float*)d_in[5];
    const float* bg  = (const float*)d_in[6];
    const float* bW1 = (const float*)d_in[7];
    const float* bb1 = (const float*)d_in[8];
    const float* bW2 = (const float*)d_in[9];
    const float* bb2 = (const float*)d_in[10];
    const float* bW3 = (const float*)d_in[11];
    const float* bb3 = (const float*)d_in[12];
    const float* bW4 = (const float*)d_in[13];
    const float* bb4 = (const float*)d_in[14];
    const float* sW1 = (const float*)d_in[15];
    const float* sb1 = (const float*)d_in[16];
    const float* sW2 = (const float*)d_in[17];
    const float* sb2 = (const float*)d_in[18];
    const float* sW3 = (const float*)d_in[19];
    const float* sb3 = (const float*)d_in[20];
    const float* sW4 = (const float*)d_in[21];
    const float* sb4 = (const float*)d_in[22];

    float* outF = (float*)d_out;
    char*  wsb  = (char*)d_ws;

    __bf16* featH  = (__bf16*)(wsb + OFF_FEATBF);
    __bf16* wvH    = (__bf16*)(wsb + OFF_WV);
    __bf16* wuH    = (__bf16*)(wsb + OFF_WU);
    __bf16* w1H    = (__bf16*)(wsb + OFF_W1);
    float*  wsA    = (float*)(wsb + OFF_A);
    float*  wsW    = (float*)(wsb + OFF_WSM);
    float*  wsBag  = (float*)(wsb + OFF_BAG);
    float*  wsO2   = (float*)(wsb + OFF_O2);
    float*  wsCent = (float*)(wsb + OFF_CENT);
    int*    wsLbl  = (int*)(wsb + OFF_LBL);
    int*    wsCnt  = (int*)(wsb + OFF_CNT);
    float*  wsC2   = (float*)(wsb + OFF_C2);
    float*  wsH1b  = (float*)(wsb + OFF_H1B);

    float* featOut = outF + O_FEAT;
    float* lblOut  = outF + O_LBL;
    float* segOut  = outF + O_SEG;
    float* outOut  = outF + O_OUT;
    float* AtOut   = outF + O_AT;

    k0_prep<<<ROWS, 256, 0, stream>>>(x, featOut, featH);
    k1_wconv<<<(Dd * N1) / (256 * 4), 256, 0, stream>>>(Wv, Wu, sW1, wvH, wuH, w1H);
    k2_attn_seg<<<ROWS / MROW, 512, 0, stream>>>(featH, wvH, wuH, w1H,
                                                 bv, bu, Wg, bg, sb1,
                                                 sW2, sb2, sW3, sb3, sW4, sb4,
                                                 segOut, AtOut, wsA);
    k3_softmax_max<<<Bb, 1024, 0, stream>>>(wsA, segOut, wsW, wsO2);
    k4_bag<<<dim3(Dd / 512, Bb), 512, 0, stream>>>(featH, wsW, wsBag);
    k5a_bagL1<<<dim3(N1 / 128, Bb), 128, 0, stream>>>(wsBag, bW1, bb1, wsH1b);
    k5b_bagtail<<<1, 512, 0, stream>>>(wsH1b, bW2, bb2, bW3, bb3, bW4, bb4,
                                       wsO2, outOut);

    // iterative whole-GPU kmeans: init, 25x(prep,assign,update), final (prep,assign)
    k6a_init<<<Bb, 512, 0, stream>>>(featH, wsCent);
    for (int it = 0; it < 26; ++it) {
        k6b_prep<<<Bb, 256, 0, stream>>>(wsCent, wsC2, wsCnt);
        k6c_assign<<<dim3(Tt / 32, Bb), 256, 0, stream>>>(featH, wsCent, wsC2,
                                                          wsLbl, wsCnt, lblOut);
        if (it < 25)
            k6d_update<<<dim3(Dd / 512, Bb), 512, 0, stream>>>(featH, wsLbl, wsCnt, wsCent);
    }
}
